// FalconH1SparseMoeBlock_61203283968371
// MI455X (gfx1250) — compile-verified, matrix operand order checked by reading, not run on hardware
//
#include <hip/hip_runtime.h>

#define NUM_EXPERTS 8
#define HIDDEN 2048
#define INTER 4096
#define TOKENS 8192
#define GATE_MULT 0.5f
#define DOWN_MULT 0.25f

// ---- types for WMMA ----
typedef __bf16 bf16;
typedef __attribute__((ext_vector_type(16))) __bf16 v16bf;
typedef __attribute__((ext_vector_type(8)))  __bf16 v8bf;
typedef __attribute__((ext_vector_type(8)))  float   v8f;
typedef __attribute__((ext_vector_type(4)))  float   v4f;
typedef __attribute__((ext_vector_type(4)))  int     i4;

union ABFrag { v16bf v; v8bf h[2]; };

// native RNE f32->bf16 (lowers to hardware cvt on gfx1250)
__device__ __forceinline__ bf16 f2bf(float f) { return (bf16)f; }

// pack two f32 into a dword of two bf16 (lo = first, hi = second)
__device__ __forceinline__ unsigned pk2bf(float lo, float hi) {
  unsigned short l = __builtin_bit_cast(unsigned short, (bf16)lo);
  unsigned short h = __builtin_bit_cast(unsigned short, (bf16)hi);
  return ((unsigned)h << 16) | (unsigned)l;
}

// 16-byte async global->LDS copy (gfx1250 GLOBAL_LOAD_ASYNC_TO_LDS_B128)
__device__ __forceinline__ void async_copy16(const bf16* g, bf16* l) {
  __builtin_amdgcn_global_load_async_to_lds_b128(
      (__attribute__((address_space(1))) i4*)(i4*)(bf16*)g,
      (__attribute__((address_space(3))) i4*)(i4*)l, 0, 0);
}
__device__ __forceinline__ void wait_async0() {
  asm volatile("s_wait_asynccnt 0x0" ::: "memory");
}

__device__ __forceinline__ float fast_sigmoid(float z) {
  return __builtin_amdgcn_rcpf(1.f + __expf(-z));
}

// ---------------------------------------------------------------------------
// Router: logits = x @ w_router, softmax over 8, top-2 (no renorm),
// write dense combine weights w_comb[T][8] (0 for unselected experts).
// ---------------------------------------------------------------------------
__global__ __launch_bounds__(256)
void moe_router_kernel(const float* __restrict__ x,
                       const float* __restrict__ w_router,
                       float* __restrict__ w_comb) {
  int t = blockIdx.x * blockDim.x + threadIdx.x;
  if (t >= TOKENS) return;
  float acc[NUM_EXPERTS];
#pragma unroll
  for (int e = 0; e < NUM_EXPERTS; ++e) acc[e] = 0.f;
  const float* xr = x + (size_t)t * HIDDEN;
  for (int h = 0; h < HIDDEN; ++h) {
    float xv = xr[h];
    const float* wr = w_router + (size_t)h * NUM_EXPERTS;
#pragma unroll
    for (int e = 0; e < NUM_EXPERTS; ++e) acc[e] += xv * wr[e];
  }
  float mx = acc[0];
#pragma unroll
  for (int e = 1; e < NUM_EXPERTS; ++e) mx = fmaxf(mx, acc[e]);
  float p[NUM_EXPERTS];
  float s = 0.f;
#pragma unroll
  for (int e = 0; e < NUM_EXPERTS; ++e) { p[e] = __expf(acc[e] - mx); s += p[e]; }
  float inv = __builtin_amdgcn_rcpf(s);
#pragma unroll
  for (int e = 0; e < NUM_EXPERTS; ++e) p[e] *= inv;
  int i1 = 0;
#pragma unroll
  for (int e = 1; e < NUM_EXPERTS; ++e) if (p[e] > p[i1]) i1 = e;
  int i2 = (i1 == 0) ? 1 : 0;
#pragma unroll
  for (int e = 0; e < NUM_EXPERTS; ++e)
    if (e != i1 && e != i2 && p[e] > p[i2]) i2 = e;
  float* wc = w_comb + (size_t)t * NUM_EXPERTS;
#pragma unroll
  for (int e = 0; e < NUM_EXPERTS; ++e)
    wc[e] = (e == i1 || e == i2) ? p[e] : 0.f;
}

// ---------------------------------------------------------------------------
// One-time x (fp32) -> xb (bf16). Lets all 8 expert GEMMs stream A as bf16
// and fill A tiles with pure async byte-copies (no VALU conversion).
// ---------------------------------------------------------------------------
__global__ __launch_bounds__(256)
void x_to_bf16_kernel(const float* __restrict__ x, bf16* __restrict__ xb) {
  size_t i = ((size_t)blockIdx.x * blockDim.x + threadIdx.x) * 8;
  v4f a0 = *(const v4f*)(x + i);
  v4f a1 = *(const v4f*)(x + i + 4);
  v8bf v;
#pragma unroll
  for (int k = 0; k < 4; ++k) { v[k] = f2bf(a0[k]); v[4 + k] = f2bf(a1[k]); }
  *(v8bf*)(xb + i) = v;
}

// ---------------------------------------------------------------------------
// GEMM 1 (per expert): act[t][i] = silu(0.5 * x@wg) * (x@wu), bf16 output.
// 64x64 tile / block, 256 threads (8 wave32), BK=32, bf16 WMMA 16x16x32.
// A tile: async global->LDS copy. B tiles: fp32 load, packed bf16 cvt,
// transposed [n][k] store as b32 pairs.
// ---------------------------------------------------------------------------
#define LD 40  // LDS row stride in bf16 elements (32 + 8 pad; 80B keeps 16B align)

__global__ __launch_bounds__(256)
void moe_gateup_kernel(const bf16* __restrict__ xb,
                       const float* __restrict__ wg,   // [HIDDEN][INTER] slice
                       const float* __restrict__ wu,   // [HIDDEN][INTER] slice
                       bf16* __restrict__ act) {
  __shared__ bf16 lA[64 * LD];
  __shared__ bf16 lG[64 * LD];
  __shared__ bf16 lU[64 * LD];

  const int tid  = threadIdx.x;
  const int lane = tid & 31;
  const int wid  = tid >> 5;
  const int wm   = wid & 3;       // 4 wave rows of 16
  const int wn   = wid >> 2;      // 2 wave cols of 32
  const int bM   = blockIdx.y * 64;
  const int bN   = blockIdx.x * 64;

  v8f accG0 = {}, accG1 = {}, accU0 = {}, accU1 = {};

  const int ar = tid >> 2;            // A: row in tile 0..63
  const int ak = (tid & 3) * 8;       // A: k segment (x8 bf16 = 16B)
  const int kp = tid >> 4;            // B: k-row pair 0..15 -> rows 2kp,2kp+1
  const int n0 = (tid & 15) * 4;      // B: n segment of 4

  const int arow = wm * 16 + (lane & 15);
  const int kg   = (lane >> 4) * 8;   // K-half selector (ISA 16-bit A layout)
  const int bc0  = wn * 32 + (lane & 15);
  const int bc1  = bc0 + 16;

  for (int k0 = 0; k0 < HIDDEN; k0 += 32) {
    // --- A tile: pure async copy (already bf16) ---
    async_copy16(xb + (size_t)(bM + ar) * HIDDEN + (k0 + ak), &lA[ar * LD + ak]);

    // --- B tiles: fp32 -> bf16, transposed [n][k], packed dword stores ---
    const float* gp0 = wg + (size_t)(k0 + 2 * kp)     * INTER + (bN + n0);
    const float* gp1 = wg + (size_t)(k0 + 2 * kp + 1) * INTER + (bN + n0);
    const float* up0 = wu + (size_t)(k0 + 2 * kp)     * INTER + (bN + n0);
    const float* up1 = wu + (size_t)(k0 + 2 * kp + 1) * INTER + (bN + n0);
    v4f g0 = *(const v4f*)gp0, g1 = *(const v4f*)gp1;
    v4f u0 = *(const v4f*)up0, u1 = *(const v4f*)up1;
#pragma unroll
    for (int i = 0; i < 4; ++i) {
      *(unsigned*)&lG[(n0 + i) * LD + 2 * kp] = pk2bf(g0[i], g1[i]);
      *(unsigned*)&lU[(n0 + i) * LD + 2 * kp] = pk2bf(u0[i], u1[i]);
    }
    wait_async0();
    __syncthreads();

    // --- fragments ---
    ABFrag a, fg0, fg1, fu0, fu1;
    a.h[0]   = *(v8bf*)&lA[arow * LD + kg];
    a.h[1]   = *(v8bf*)&lA[arow * LD + 16 + kg];
    fg0.h[0] = *(v8bf*)&lG[bc0 * LD + kg];
    fg0.h[1] = *(v8bf*)&lG[bc0 * LD + 16 + kg];
    fg1.h[0] = *(v8bf*)&lG[bc1 * LD + kg];
    fg1.h[1] = *(v8bf*)&lG[bc1 * LD + 16 + kg];
    fu0.h[0] = *(v8bf*)&lU[bc0 * LD + kg];
    fu0.h[1] = *(v8bf*)&lU[bc0 * LD + 16 + kg];
    fu1.h[0] = *(v8bf*)&lU[bc1 * LD + kg];
    fu1.h[1] = *(v8bf*)&lU[bc1 * LD + 16 + kg];

    accG0 = __builtin_amdgcn_wmma_f32_16x16x32_bf16(false, a.v, false, fg0.v, (short)0, accG0, false, false);
    accG1 = __builtin_amdgcn_wmma_f32_16x16x32_bf16(false, a.v, false, fg1.v, (short)0, accG1, false, false);
    accU0 = __builtin_amdgcn_wmma_f32_16x16x32_bf16(false, a.v, false, fu0.v, (short)0, accU0, false, false);
    accU1 = __builtin_amdgcn_wmma_f32_16x16x32_bf16(false, a.v, false, fu1.v, (short)0, accU1, false, false);
    __syncthreads();
  }

  // --- epilogue: silu(0.5*g) * u -> bf16 act (rcp-based sigmoid, no IEEE div) ---
#pragma unroll
  for (int r = 0; r < 8; ++r) {
    int m = bM + wm * 16 + r + ((lane >> 4) << 3);
    int n = bN + wn * 32 + (lane & 15);
    float g = accG0[r] * GATE_MULT;
    act[(size_t)m * INTER + n] = f2bf(g * fast_sigmoid(g) * accU0[r]);
    g = accG1[r] * GATE_MULT;
    act[(size_t)m * INTER + n + 16] = f2bf(g * fast_sigmoid(g) * accU1[r]);
  }
}

// ---------------------------------------------------------------------------
// GEMM 2 (per expert): out[t][h] (+)= w_comb[t][e] * 0.25 * (act @ w_down)
// e==0 initializes out (harness poisons it).
// ---------------------------------------------------------------------------
__global__ __launch_bounds__(256)
void moe_down_kernel(const bf16* __restrict__ act,
                     const float* __restrict__ wd,      // [INTER][HIDDEN] slice
                     const float* __restrict__ w_comb,
                     float* __restrict__ out, int e) {
  __shared__ bf16 lA[64 * LD];
  __shared__ bf16 lB[64 * LD];

  const int tid  = threadIdx.x;
  const int lane = tid & 31;
  const int wid  = tid >> 5;
  const int wm   = wid & 3;
  const int wn   = wid >> 2;
  const int bM   = blockIdx.y * 64;
  const int bN   = blockIdx.x * 64;

  v8f acc0 = {}, acc1 = {};

  const int ar = tid >> 2;
  const int ak = (tid & 3) * 8;
  const int kp = tid >> 4;
  const int n0 = (tid & 15) * 4;

  const int arow = wm * 16 + (lane & 15);
  const int kg   = (lane >> 4) * 8;
  const int bc0  = wn * 32 + (lane & 15);
  const int bc1  = bc0 + 16;

  for (int k0 = 0; k0 < INTER; k0 += 32) {
    // A tile (bf16): async copy
    async_copy16(act + (size_t)(bM + ar) * INTER + (k0 + ak), &lA[ar * LD + ak]);

    // B tile fp32 -> bf16, transposed [n][k]
    const float* bp0 = wd + (size_t)(k0 + 2 * kp)     * HIDDEN + (bN + n0);
    const float* bp1 = wd + (size_t)(k0 + 2 * kp + 1) * HIDDEN + (bN + n0);
    v4f b0 = *(const v4f*)bp0, b1 = *(const v4f*)bp1;
#pragma unroll
    for (int i = 0; i < 4; ++i) {
      *(unsigned*)&lB[(n0 + i) * LD + 2 * kp] = pk2bf(b0[i], b1[i]);
    }
    wait_async0();
    __syncthreads();

    ABFrag a, f0, f1;
    a.h[0]  = *(v8bf*)&lA[arow * LD + kg];
    a.h[1]  = *(v8bf*)&lA[arow * LD + 16 + kg];
    f0.h[0] = *(v8bf*)&lB[bc0 * LD + kg];
    f0.h[1] = *(v8bf*)&lB[bc0 * LD + 16 + kg];
    f1.h[0] = *(v8bf*)&lB[bc1 * LD + kg];
    f1.h[1] = *(v8bf*)&lB[bc1 * LD + 16 + kg];

    acc0 = __builtin_amdgcn_wmma_f32_16x16x32_bf16(false, a.v, false, f0.v, (short)0, acc0, false, false);
    acc1 = __builtin_amdgcn_wmma_f32_16x16x32_bf16(false, a.v, false, f1.v, (short)0, acc1, false, false);
    __syncthreads();
  }

#pragma unroll
  for (int r = 0; r < 8; ++r) {
    int m = bM + wm * 16 + r + ((lane >> 4) << 3);
    int n = bN + wn * 32 + (lane & 15);
    float coef = DOWN_MULT * w_comb[(size_t)m * NUM_EXPERTS + e];
    size_t o = (size_t)m * HIDDEN + n;
    float p0 = (e == 0) ? 0.f : out[o];
    out[o] = p0 + coef * acc0[r];
    float p1 = (e == 0) ? 0.f : out[o + 16];
    out[o + 16] = p1 + coef * acc1[r];
  }
}

// ---------------------------------------------------------------------------
extern "C" void kernel_launch(void* const* d_in, const int* in_sizes, int n_in,
                              void* d_out, int out_size, void* d_ws, size_t ws_size,
                              hipStream_t stream) {
  (void)in_sizes; (void)n_in; (void)out_size; (void)ws_size;
  const float* x        = (const float*)d_in[0];   // [4,2048,2048]
  const float* w_router = (const float*)d_in[1];   // [2048,8]
  const float* w_gate   = (const float*)d_in[2];   // [8,2048,4096]
  const float* w_up     = (const float*)d_in[3];   // [8,2048,4096]
  const float* w_down   = (const float*)d_in[4];   // [8,4096,2048]
  float* out = (float*)d_out;

  char* ws = (char*)d_ws;
  float* w_comb = (float*)ws;                                  // 8192*8 f32
  size_t off = ((size_t)TOKENS * NUM_EXPERTS * sizeof(float) + 255) & ~(size_t)255;
  bf16* xb = (bf16*)(ws + off);                                // 32 MiB
  off += (size_t)TOKENS * HIDDEN * sizeof(bf16);
  bf16* act = (bf16*)(ws + off);                               // 64 MiB

  moe_router_kernel<<<TOKENS / 256, 256, 0, stream>>>(x, w_router, w_comb);
  x_to_bf16_kernel<<<(TOKENS * (size_t)HIDDEN) / (256 * 8), 256, 0, stream>>>(x, xb);

  dim3 g1(INTER / 64, TOKENS / 64);
  dim3 g2(HIDDEN / 64, TOKENS / 64);
  for (int e = 0; e < NUM_EXPERTS; ++e) {
    const float* wg = w_gate + (size_t)e * HIDDEN * INTER;
    const float* wu = w_up   + (size_t)e * HIDDEN * INTER;
    const float* wd = w_down + (size_t)e * INTER * HIDDEN;
    moe_gateup_kernel<<<g1, 256, 0, stream>>>(xb, wg, wu, act);
    moe_down_kernel<<<g2, 256, 0, stream>>>(act, wd, w_comb, out, e);
  }
}
